// Rep_24386824306842
// MI455X (gfx1250) — compile-verified, matrix operand order checked by reading, not run on hardware
//
#include <hip/hip_runtime.h>

#define DIMD 64
#define NNODES 50000
#define NEDGES 100000
#define NGRAPHS 2048

typedef __attribute__((ext_vector_type(16))) __bf16 v16bf;
typedef __attribute__((ext_vector_type(8)))  float  v8f;

// ---------------------------------------------------------------- helpers
__device__ __forceinline__ unsigned short f2bfbits(float f) {
  unsigned u = __builtin_bit_cast(unsigned, f);
  unsigned r = u + 0x7FFFu + ((u >> 16) & 1u);   // round-to-nearest-even
  return (unsigned short)(r >> 16);
}
// loads bf16 fragment slice: p[0..7] and p[16..23] (two 16B runs)
__device__ __forceinline__ v16bf load_frag_bf16(const unsigned short* p) {
  union { uint4 q[2]; v16bf v; } u;
  u.q[0] = *(const uint4*)(p);
  u.q[1] = *(const uint4*)(p + 16);
  return u.v;
}
__device__ __forceinline__ float sigmoidf(float x) { return 1.0f / (1.0f + __expf(-x)); }
__device__ __forceinline__ unsigned f2ord(float f) {
  unsigned u = __builtin_bit_cast(unsigned, f);
  return (u & 0x80000000u) ? ~u : (u | 0x80000000u);
}
__device__ __forceinline__ float ord2f(unsigned v) {
  unsigned u = (v & 0x80000000u) ? (v & 0x7FFFFFFFu) : ~v;
  return __builtin_bit_cast(float, u);
}

// ---------------------------------------------------------------- tiny kernels
__global__ void zero_f32(float* p, int n) {
  int i = blockIdx.x * 256 + threadIdx.x;
  if (i < n) p[i] = 0.0f;
}
__global__ void cvt_to_bf16(const float* __restrict__ in, unsigned short* __restrict__ out, int n) {
  int i = blockIdx.x * 256 + threadIdx.x;
  if (i < n) out[i] = f2bfbits(in[i]);
}
// h = relu(x @ lin0_w^T + b)   (K=11 -> VALU, tiny); writes f32 + bf16 shadow
__global__ void lin0_relu(const float* __restrict__ x, const float* __restrict__ w,
                          const float* __restrict__ b, float* __restrict__ h,
                          unsigned short* __restrict__ hbf) {
  int i = blockIdx.x * 256 + threadIdx.x;
  if (i >= NNODES * DIMD) return;
  int n = i >> 6, o = i & 63;
  float acc = b[o];
#pragma unroll
  for (int k = 0; k < 11; ++k) acc += x[n * 11 + k] * w[o * 11 + k];
  float v = fmaxf(acc, 0.0f);
  h[i] = v;
  hbf[i] = f2bfbits(v);
}
// he = relu(edge_attr @ mlp_w1^T + b1)  stored bf16 (K=5 -> VALU, tiny)
__global__ void edge_mlp1(const float* __restrict__ ea, const float* __restrict__ w,
                          const float* __restrict__ b, unsigned short* __restrict__ he) {
  int i = blockIdx.x * 256 + threadIdx.x;
  if (i >= NEDGES * 128) return;
  int e = i >> 7, j = i & 127;
  float acc = b[j];
#pragma unroll
  for (int k = 0; k < 5; ++k) acc += ea[e * 5 + k] * w[j * 5 + k];
  he[i] = f2bfbits(fmaxf(acc, 0.0f));
}
__global__ void edge_count(const int* __restrict__ dst, float* __restrict__ cnt) {
  int e = blockIdx.x * 256 + threadIdx.x;
  if (e < NEDGES) atomicAdd(&cnt[dst[e]], 1.0f);
}
// m = relu(agg/max(cnt,1) + (h@root_w^T + conv_bias)) -> bf16 (only GEMM consumes m)
__global__ void node_m(const float* __restrict__ agg, const float* __restrict__ cnt,
                       const float* __restrict__ troot, unsigned short* __restrict__ mbf) {
  int i = blockIdx.x * 256 + threadIdx.x;
  if (i >= NNODES * DIMD) return;
  float cc = fmaxf(cnt[i >> 6], 1.0f);
  mbf[i] = f2bfbits(fmaxf(agg[i] / cc + troot[i], 0.0f));
}
__global__ void gru_update(const float* __restrict__ gi, const float* __restrict__ gh,
                           float* __restrict__ h, unsigned short* __restrict__ hbf) {
  int i = blockIdx.x * 256 + threadIdx.x;
  if (i >= NNODES * DIMD) return;
  int n = i >> 6, o = i & 63;
  size_t base = (size_t)n * 192;
  float r = sigmoidf(gi[base + o] + gh[base + o]);
  float z = sigmoidf(gi[base + 64 + o] + gh[base + 64 + o]);
  float ng = tanhf(gi[base + 128 + o] + r * gh[base + 128 + o]);
  float v = (1.0f - z) * ng + z * h[i];
  h[i] = v;
  hbf[i] = f2bfbits(v);
}
__global__ void lstm_update(const float* __restrict__ g1, const float* __restrict__ g2,
                            float* __restrict__ cl, float* __restrict__ hl,
                            unsigned short* __restrict__ hlbf) {
  int i = blockIdx.x * 256 + threadIdx.x;
  if (i >= NGRAPHS * DIMD) return;
  int b = i >> 6, o = i & 63;
  size_t base = (size_t)b * 256;
  float ig = sigmoidf(g1[base + o]       + g2[base + o]);
  float fg = sigmoidf(g1[base + 64 + o]  + g2[base + 64 + o]);
  float gg = tanhf  (g1[base + 128 + o] + g2[base + 128 + o]);
  float og = sigmoidf(g1[base + 192 + o] + g2[base + 192 + o]);
  float c = fg * cl[i] + ig * gg;
  cl[i] = c;
  float v = og * tanhf(c);
  hl[i] = v;
  hlbf[i] = f2bfbits(v);
}
__global__ void att_e(const float* __restrict__ h, const float* __restrict__ hl,
                      const int* __restrict__ batch, float* __restrict__ e,
                      unsigned* __restrict__ emax) {
  int n = blockIdx.x * 256 + threadIdx.x;
  if (n >= NNODES) return;
  int b = batch[n];
  float d = 0.0f;
#pragma unroll 8
  for (int o = 0; o < DIMD; ++o) d += h[(size_t)n * DIMD + o] * hl[(size_t)b * DIMD + o];
  e[n] = d;
  atomicMax(&emax[b], f2ord(d));
}
__global__ void att_a(const float* __restrict__ e, const unsigned* __restrict__ emax,
                      const int* __restrict__ batch, float* __restrict__ a,
                      float* __restrict__ asum) {
  int n = blockIdx.x * 256 + threadIdx.x;
  if (n >= NNODES) return;
  int b = batch[n];
  float v = __expf(e[n] - ord2f(emax[b]));
  a[n] = v;
  atomicAdd(&asum[b], v);
}
__global__ void rpool_k(const float* __restrict__ a, const float* __restrict__ asum,
                        const float* __restrict__ h, const int* __restrict__ batch,
                        float* __restrict__ rpool) {
  int i = blockIdx.x * 256 + threadIdx.x;
  if (i >= NNODES * DIMD) return;
  int n = i >> 6, o = i & 63;
  int b = batch[n];
  float w = a[n] / asum[b];
  atomicAdd(&rpool[(size_t)b * DIMD + o], w * h[i]);
}
// q_star = concat(hl, rpool) stored bf16 (only GEMMs consume it)
__global__ void qstar_k(const float* __restrict__ hl, const float* __restrict__ rpool,
                        unsigned short* __restrict__ qstarbf) {
  int i = blockIdx.x * 256 + threadIdx.x;
  if (i >= NGRAPHS * 2 * DIMD) return;
  int b = i >> 7, j = i & 127;
  float v = (j < 64) ? hl[(size_t)b * 64 + j] : rpool[(size_t)b * 64 + j - 64];
  qstarbf[i] = f2bfbits(v);
}

// ---------------------------------------------------------------- generic Y = act(X @ W^T + bias), WMMA bf16
// X: [M,K] bf16, W: [Nout,K] bf16, Y: [M,Nout] f32. K % 32 == 0; grid=(M/16, Nout/16), block=32 (one wave).
__global__ __launch_bounds__(32) void gemm_xwt_wmma(
    const unsigned short* __restrict__ X, const unsigned short* __restrict__ W,
    const float* __restrict__ bias, float* __restrict__ Y,
    int K, int Nout, int relu) {
  const int m0 = blockIdx.x * 16, n0 = blockIdx.y * 16;
  const int lane = threadIdx.x & 31;
  const int mrow = lane & 15, hi = lane >> 4;
  v8f c = {};
  for (int k0 = 0; k0 < K; k0 += 32) {
    v16bf a = load_frag_bf16(X + (size_t)(m0 + mrow) * K + k0 + hi * 8);
    v16bf b = load_frag_bf16(W + (size_t)(n0 + mrow) * K + k0 + hi * 8);
    c = __builtin_amdgcn_wmma_f32_16x16x32_bf16(false, a, false, b, (short)0, c, false, false);
  }
  const float bb = bias ? bias[n0 + mrow] : 0.0f;
#pragma unroll
  for (int r = 0; r < 8; ++r) {
    float v = c[r] + bb;
    if (relu) v = fmaxf(v, 0.0f);
    Y[(size_t)(m0 + r + hi * 8) * Nout + n0 + mrow] = v;
  }
}

// ---------------------------------------------------------------- fused NNConv message pass
// Per block: 16 edges. Stage he tile (16x256B, contiguous) into LDS via the CDNA5 async
// global->LDS engine, recompute We tile = he(16x128) @ w2^T(128x4096) via WMMA (L2-resident
// inputs), contract against gathered x_src in-register, reduce across waves with ds_add_f32,
// scatter-add to agg[dst].
__global__ __launch_bounds__(256) void nnconv_msg_wmma(
    const unsigned short* __restrict__ he,  // [E,128] bf16
    const unsigned short* __restrict__ w2,  // [4096,128] bf16
    const float* __restrict__ b2,           // [4096]
    const float* __restrict__ h,            // [N,64]
    const int* __restrict__ src, const int* __restrict__ dst,
    float* __restrict__ agg) {
  __shared__ __align__(16) unsigned short he_s[16 * 128];
  __shared__ float xs[16 * 64];
  __shared__ float msg_s[16 * 64];
  const int e0 = blockIdx.x * 16;
  const int t = threadIdx.x;

  // async copy of the 4KB he tile: 256 lanes x 16B, global -> LDS (ASYNCcnt path)
  {
    unsigned lds_off = (unsigned)(uintptr_t)(&he_s[0]) + (unsigned)t * 16u;
    unsigned long long ga =
        (unsigned long long)(uintptr_t)(he + (size_t)e0 * 128) + (unsigned long long)t * 16u;
    asm volatile("global_load_async_to_lds_b128 %0, %1, off"
                 :: "v"(lds_off), "v"(ga) : "memory");
  }
  for (int idx = t; idx < 16 * 64; idx += 256) {
    int m = idx >> 6, o = idx & 63;
    xs[idx] = h[(size_t)src[e0 + m] * 64 + o];
    msg_s[idx] = 0.0f;
  }
  asm volatile("s_wait_asynccnt 0" ::: "memory");
  __syncthreads();

  const int wave = t >> 5, lane = t & 31;
  const int mrow = lane & 15, hi = lane >> 4;

  v16bf afrag[4];
#pragma unroll
  for (int kk = 0; kk < 4; ++kk)
    afrag[kk] = load_frag_bf16(&he_s[mrow * 128 + kk * 32 + hi * 8]);

  float macc[8][4];
#pragma unroll
  for (int r = 0; r < 8; ++r)
#pragma unroll
    for (int q = 0; q < 4; ++q) macc[r][q] = 0.0f;

  // wave handles n-tiles [wave*32, wave*32+32): col0 = tt*16, fi = col0/64, o0 = col0%64
  for (int s = 0; s < 32; ++s) {
    const int tt = wave * 32 + s;
    const int col0 = tt * 16;
    const int fi = tt >> 2;   // input-feature index (fixed per 16-wide tile)
    const int q = tt & 3;     // o0 / 16
    const unsigned short* wr = w2 + (size_t)(col0 + mrow) * 128 + hi * 8;
    if (s + 1 < 32)
      __builtin_prefetch(w2 + (size_t)(col0 + 16 + mrow) * 128, 0, 3);
    v8f c = {};
#pragma unroll
    for (int kk = 0; kk < 4; ++kk) {
      v16bf b = load_frag_bf16(wr + kk * 32);
      c = __builtin_amdgcn_wmma_f32_16x16x32_bf16(false, afrag[kk], false, b, (short)0, c, false, false);
    }
    const float bb = b2[col0 + mrow];  // bias column for this lane
#pragma unroll
    for (int r = 0; r < 8; ++r) {      // c[r] = We[e0 + r + hi*8, col0 + mrow]
      const float xv = xs[(r + hi * 8) * 64 + fi];
      macc[r][q] += xv * (c[r] + bb);
    }
  }
#pragma unroll
  for (int r = 0; r < 8; ++r)
#pragma unroll
    for (int q = 0; q < 4; ++q)
      atomicAdd(&msg_s[(r + hi * 8) * 64 + q * 16 + mrow], macc[r][q]);  // ds_add_f32
  __syncthreads();

  for (int idx = t; idx < 16 * 64; idx += 256) {
    int m = idx >> 6, o = idx & 63;
    atomicAdd(&agg[(size_t)dst[e0 + m] * 64 + o], msg_s[idx]);
  }
}

// ---------------------------------------------------------------- host
extern "C" void kernel_launch(void* const* d_in, const int* in_sizes, int n_in,
                              void* d_out, int out_size, void* d_ws, size_t ws_size,
                              hipStream_t stream) {
  const float* x         = (const float*)d_in[0];
  const float* edge_attr = (const float*)d_in[1];
  const int*   eidx      = (const int*)d_in[2];
  const int*   batch     = (const int*)d_in[3];
  const float* lin0_w    = (const float*)d_in[5];
  const float* lin0_b    = (const float*)d_in[6];
  const float* mlp_w1    = (const float*)d_in[7];
  const float* mlp_b1    = (const float*)d_in[8];
  const float* mlp_w2    = (const float*)d_in[9];
  const float* mlp_b2    = (const float*)d_in[10];
  const float* root_w    = (const float*)d_in[11];
  const float* conv_bias = (const float*)d_in[12];
  const float* gru_w_ih  = (const float*)d_in[13];
  const float* gru_w_hh  = (const float*)d_in[14];
  const float* gru_b_ih  = (const float*)d_in[15];
  const float* gru_b_hh  = (const float*)d_in[16];
  const float* lstm_w_ih = (const float*)d_in[17];
  const float* lstm_w_hh = (const float*)d_in[18];
  const float* lstm_b_ih = (const float*)d_in[19];
  const float* lstm_b_hh = (const float*)d_in[20];
  const float* lin1_w    = (const float*)d_in[21];
  const float* lin1_b    = (const float*)d_in[22];
  const int* src = eidx;
  const int* dst = eidx + NEDGES;

  // workspace allocator (256B aligned)
  size_t off = 0;
  auto alloc = [&](size_t bytes) -> void* {
    off = (off + 255) & ~(size_t)255;
    void* p = (char*)d_ws + off;
    off += bytes;
    return p;
  };
  unsigned short* w2bf   = (unsigned short*)alloc((size_t)4096 * 128 * 2);
  unsigned short* rootbf = (unsigned short*)alloc(64 * 64 * 2);
  unsigned short* gihbf  = (unsigned short*)alloc(192 * 64 * 2);
  unsigned short* ghhbf  = (unsigned short*)alloc(192 * 64 * 2);
  unsigned short* lihbf  = (unsigned short*)alloc(256 * 128 * 2);
  unsigned short* lhhbf  = (unsigned short*)alloc(256 * 64 * 2);
  unsigned short* l1bf   = (unsigned short*)alloc(64 * 128 * 2);
  unsigned short* hebf   = (unsigned short*)alloc((size_t)NEDGES * 128 * 2);
  float* hbuf  = (float*)alloc((size_t)NNODES * 64 * 4);
  unsigned short* hbf = (unsigned short*)alloc((size_t)NNODES * 64 * 2);
  unsigned short* mbf = (unsigned short*)alloc((size_t)NNODES * 64 * 2);
  float* agg   = (float*)alloc((size_t)NNODES * 64 * 4);
  float* troot = (float*)alloc((size_t)NNODES * 64 * 4);
  float* gi    = (float*)alloc((size_t)NNODES * 192 * 4);
  float* gh    = (float*)alloc((size_t)NNODES * 192 * 4);
  float* cnt   = (float*)alloc((size_t)NNODES * 4);
  unsigned short* qstarbf = (unsigned short*)alloc((size_t)NGRAPHS * 128 * 2);
  float* hl    = (float*)alloc((size_t)NGRAPHS * 64 * 4);
  unsigned short* hlbf = (unsigned short*)alloc((size_t)NGRAPHS * 64 * 2);
  float* cl    = (float*)alloc((size_t)NGRAPHS * 64 * 4);
  float* g1buf = (float*)alloc((size_t)NGRAPHS * 256 * 4);
  float* g2buf = (float*)alloc((size_t)NGRAPHS * 256 * 4);
  float* ebuf  = (float*)alloc((size_t)NNODES * 4);
  float* abuf  = (float*)alloc((size_t)NNODES * 4);
  float* emax  = (float*)alloc((size_t)NGRAPHS * 4);
  float* asum  = (float*)alloc((size_t)NGRAPHS * 4);
  float* rpool = (float*)alloc((size_t)NGRAPHS * 64 * 4);
  (void)in_sizes; (void)n_in; (void)out_size; (void)ws_size;

  auto G = [](int n) { return dim3((unsigned)((n + 255) / 256)); };

  // --- weight conversions to bf16 (once per call)
  cvt_to_bf16<<<G(4096 * 128), 256, 0, stream>>>(mlp_w2, w2bf, 4096 * 128);
  cvt_to_bf16<<<G(64 * 64),   256, 0, stream>>>(root_w, rootbf, 64 * 64);
  cvt_to_bf16<<<G(192 * 64),  256, 0, stream>>>(gru_w_ih, gihbf, 192 * 64);
  cvt_to_bf16<<<G(192 * 64),  256, 0, stream>>>(gru_w_hh, ghhbf, 192 * 64);
  cvt_to_bf16<<<G(256 * 128), 256, 0, stream>>>(lstm_w_ih, lihbf, 256 * 128);
  cvt_to_bf16<<<G(256 * 64),  256, 0, stream>>>(lstm_w_hh, lhhbf, 256 * 64);
  cvt_to_bf16<<<G(64 * 128),  256, 0, stream>>>(lin1_w, l1bf, 64 * 128);

  // --- input encoders
  lin0_relu<<<G(NNODES * 64), 256, 0, stream>>>(x, lin0_w, lin0_b, hbuf, hbf);
  edge_mlp1<<<G(NEDGES * 128), 256, 0, stream>>>(edge_attr, mlp_w1, mlp_b1, hebf);
  zero_f32<<<G(NNODES), 256, 0, stream>>>(cnt, NNODES);
  edge_count<<<G(NEDGES), 256, 0, stream>>>(dst, cnt);

  // --- 3 message-passing + GRU iterations
  for (int it = 0; it < 3; ++it) {
    zero_f32<<<G(NNODES * 64), 256, 0, stream>>>(agg, NNODES * 64);
    nnconv_msg_wmma<<<dim3(NEDGES / 16), 256, 0, stream>>>(hebf, w2bf, mlp_b2, hbuf, src, dst, agg);
    gemm_xwt_wmma<<<dim3(NNODES / 16, 4), 32, 0, stream>>>(hbf, rootbf, conv_bias, troot, 64, 64, 0);
    node_m<<<G(NNODES * 64), 256, 0, stream>>>(agg, cnt, troot, mbf);
    gemm_xwt_wmma<<<dim3(NNODES / 16, 12), 32, 0, stream>>>(mbf, gihbf, gru_b_ih, gi, 64, 192, 0);
    gemm_xwt_wmma<<<dim3(NNODES / 16, 12), 32, 0, stream>>>(hbf, ghhbf, gru_b_hh, gh, 64, 192, 0);
    gru_update<<<G(NNODES * 64), 256, 0, stream>>>(gi, gh, hbuf, hbf);
  }

  // --- Set2Set, 3 processing steps
  zero_f32<<<G(NGRAPHS * 64), 256, 0, stream>>>((float*)qstarbf, NGRAPHS * 64);  // bf16 zeros
  zero_f32<<<G(NGRAPHS * 64), 256, 0, stream>>>(hl, NGRAPHS * 64);
  zero_f32<<<G(NGRAPHS * 32), 256, 0, stream>>>((float*)hlbf, NGRAPHS * 32);     // bf16 zeros
  zero_f32<<<G(NGRAPHS * 64), 256, 0, stream>>>(cl, NGRAPHS * 64);
  for (int t = 0; t < 3; ++t) {
    gemm_xwt_wmma<<<dim3(NGRAPHS / 16, 16), 32, 0, stream>>>(qstarbf, lihbf, lstm_b_ih, g1buf, 128, 256, 0);
    gemm_xwt_wmma<<<dim3(NGRAPHS / 16, 16), 32, 0, stream>>>(hlbf, lhhbf, lstm_b_hh, g2buf, 64, 256, 0);
    lstm_update<<<G(NGRAPHS * 64), 256, 0, stream>>>(g1buf, g2buf, cl, hl, hlbf);
    zero_f32<<<G(NGRAPHS), 256, 0, stream>>>(emax, NGRAPHS);   // 0 bits == ordered -inf
    zero_f32<<<G(NGRAPHS), 256, 0, stream>>>(asum, NGRAPHS);
    zero_f32<<<G(NGRAPHS * 64), 256, 0, stream>>>(rpool, NGRAPHS * 64);
    att_e<<<G(NNODES), 256, 0, stream>>>(hbuf, hl, batch, ebuf, (unsigned*)emax);
    att_a<<<G(NNODES), 256, 0, stream>>>(ebuf, (const unsigned*)emax, batch, abuf, asum);
    rpool_k<<<G(NNODES * 64), 256, 0, stream>>>(abuf, asum, hbuf, batch, rpool);
    qstar_k<<<G(NGRAPHS * 128), 256, 0, stream>>>(hl, rpool, qstarbf);
  }

  // --- final projection: relu(q_star @ lin1_w^T + b) -> d_out [2048, 64] f32
  gemm_xwt_wmma<<<dim3(NGRAPHS / 16, 4), 32, 0, stream>>>(qstarbf, l1bf, lin1_b, (float*)d_out, 128, 64, 1);
}
